// PointNetPP_59339268161538
// MI455X (gfx1250) — matrix operand order repaired and checked by reading, not compiled, not run on hardware
//
#include <hip/hip_runtime.h>
#include <cstdint>
#include <cstddef>

typedef _Float16 h16;
typedef __attribute__((ext_vector_type(16))) _Float16 v16h;
typedef __attribute__((ext_vector_type(8)))  float    v8f;

__device__ __forceinline__ v8f wmma32(v16h a, v16h b, v8f c) {
  return __builtin_amdgcn_wmma_f32_16x16x32_f16(false, a, false, b, (short)0, c, false, false);
}

// A fragment (16x32 f16) from an LDS row-major [rows][P] f16 buffer.
// Layout: lane L holds row M=row0+(L%16); lanes<16: V0..V3 -> K 0..7, V4..V7 -> K 8..15;
// lanes>=16: +16 on the K base. Each VGPR = 2 consecutive f16 -> one 32-bit LDS read.
__device__ __forceinline__ v16h frag_a_lds(const h16* A, int P, int row0, int k0, int lane) {
  union { v16h h; unsigned u[8]; } r;
  int m  = row0 + (lane & 15);
  int hi = (lane >> 4) << 4;
  const unsigned* p0 = (const unsigned*)(A + m * P + k0 + hi);
  const unsigned* p1 = (const unsigned*)(A + m * P + k0 + 8 + hi);
#pragma unroll
  for (int v = 0; v < 4; ++v) { r.u[v] = p0[v]; r.u[4 + v] = p1[v]; }
  return r.h;
}

// B fragment (32x16 f16) from pre-transposed f16 weights Wt[N][KP] in global memory.
// lane L: column N = n0+(L%16), K base = k0 + (L/16)*16, 16 contiguous f16 = 32B load.
__device__ __forceinline__ v16h frag_b_glb(const h16* Wt, int KP, int n0, int k0, int lane) {
  return *(const v16h*)(Wt + (size_t)(n0 + (lane & 15)) * KP + k0 + ((lane >> 4) << 4));
}

// ---------------- weight prep: f32 W[Kr][Nr] -> f16 Wt[NP][KP], zero padded -------------
__global__ void prep_wt(const float* __restrict__ W, int Kr, int Nr, int KP, int NP,
                        h16* __restrict__ Wt) {
  int gid = blockIdx.x * blockDim.x + threadIdx.x;
  if (gid >= NP * KP) return;
  int n = gid / KP, k = gid % KP;
  float v = (k < Kr && n < Nr) ? W[k * Nr + n] : 0.0f;
  Wt[gid] = (h16)v;
}

// ---------------- lin_in: xb = relu(x @ W + b), x[rows][16], W[16][128] -----------------
__global__ void lin_in_kernel(const float* __restrict__ x, const float* __restrict__ W,
                              const float* __restrict__ b, float* __restrict__ xb, int rows) {
  int gid = blockIdx.x * blockDim.x + threadIdx.x;
  if (gid >= rows * 128) return;
  int r = gid / 128, c = gid % 128;
  float a = b[c];
#pragma unroll
  for (int k = 0; k < 16; ++k) a += x[r * 16 + k] * W[k * 128 + c];
  xb[gid] = fmaxf(a, 0.0f);
}

// ---------------- farthest point sampling: 1 workgroup per batch ------------------------
__global__ __launch_bounds__(256) void fps_kernel(const float* __restrict__ pos, int n, int m,
                                                  int* __restrict__ idxOut, float* __restrict__ posOut) {
  int b = blockIdx.x;
  const float* P = pos + (size_t)b * n * 3;
  int*   idx = idxOut + (size_t)b * m;
  float* Q   = posOut + (size_t)b * m * 3;
  __shared__ float dmin[4096];
  __shared__ float redv[256];
  __shared__ int   redi[256];
  __shared__ float lastp[3];
  int t = threadIdx.x;
  for (int i = t; i < n; i += 256) dmin[i] = 1e30f;
  if (t == 0) {
    idx[0] = 0; Q[0] = P[0]; Q[1] = P[1]; Q[2] = P[2];
    lastp[0] = P[0]; lastp[1] = P[1]; lastp[2] = P[2];
  }
  __syncthreads();
  for (int s = 1; s < m; ++s) {
    float lx = lastp[0], ly = lastp[1], lz = lastp[2];
    float best = -1.0f; int bi = n;
    for (int i = t; i < n; i += 256) {
      float dx = P[3 * i] - lx, dy = P[3 * i + 1] - ly, dz = P[3 * i + 2] - lz;
      float d = dx * dx + dy * dy + dz * dz;
      float dm = fminf(dmin[i], d);
      dmin[i] = dm;
      if (dm > best || (dm == best && i < bi)) { best = dm; bi = i; }
    }
    redv[t] = best; redi[t] = bi;
    __syncthreads();
    for (int off = 128; off > 0; off >>= 1) {
      if (t < off) {
        float ov = redv[t + off]; int oi = redi[t + off];
        if (ov > redv[t] || (ov == redv[t] && oi < redi[t])) { redv[t] = ov; redi[t] = oi; }
      }
      __syncthreads();
    }
    if (t == 0) {
      int w = redi[0];
      idx[s] = w;
      Q[3 * s] = P[3 * w]; Q[3 * s + 1] = P[3 * w + 1]; Q[3 * s + 2] = P[3 * w + 2];
      lastp[0] = P[3 * w]; lastp[1] = P[3 * w + 1]; lastp[2] = P[3 * w + 2];
    }
    __syncthreads();
  }
}

// ---------------- exact top-64 within radius: 1 workgroup (128 thr) per center ----------
__global__ __launch_bounds__(128) void knn_ball(const float* __restrict__ posC,
                                                const float* __restrict__ posAll,
                                                int m, int n, int* __restrict__ nbrOut) {
  int b = blockIdx.x / m, c = blockIdx.x % m;
  const float* P = posAll + (size_t)b * n * 3;
  const float* q = posC + ((size_t)b * m + c) * 3;
  __shared__ float d2[4096];
  __shared__ float redv[128];
  __shared__ int   redi[128];
  int t = threadIdx.x;
  float qx = q[0], qy = q[1], qz = q[2];
  for (int i = t; i < n; i += 128) {
    float dx = P[3 * i] - qx, dy = P[3 * i + 1] - qy, dz = P[3 * i + 2] - qz;
    float d = dx * dx + dy * dy + dz * dz;
    d2[i] = (d <= 4.0f) ? d : 1e30f;   // RADIUS^2 = 4
  }
  __syncthreads();
  int* out = nbrOut + ((size_t)b * m + c) * 64;
  for (int k = 0; k < 64; ++k) {
    float best = 1e31f; int bi = 0;
    for (int i = t; i < n; i += 128) {
      float v = d2[i];
      if (v < best || (v == best && i < bi)) { best = v; bi = i; }
    }
    redv[t] = best; redi[t] = bi;
    __syncthreads();
    for (int off = 64; off > 0; off >>= 1) {
      if (t < off) {
        float ov = redv[t + off]; int oi = redi[t + off];
        if (ov < redv[t] || (ov == redv[t] && oi < redi[t])) { redv[t] = ov; redi[t] = oi; }
      }
      __syncthreads();
    }
    if (t == 0) {
      int w = redi[0];
      out[k] = (redv[0] < 1e29f) ? w : -1;
      d2[w] = 1e31f;
    }
    __syncthreads();
  }
}

// ---------------- SA fused: gather + MLP(131->131->128) + masked max over K=64 ----------
// Tiles: M=64 (4), GEMM1 N=144 (9 tiles), K=160 (5); GEMM2 N=128 (8), K=160 (5).
__global__ __launch_bounds__(256) void sa_mlp(const float* __restrict__ featIn,
                                              const float* __restrict__ posIn,
                                              const float* __restrict__ posQ,
                                              const int* __restrict__ nbr, int m, int n,
                                              const h16* __restrict__ W1t, const float* __restrict__ b1,
                                              const float* __restrict__ g1, const float* __restrict__ be1,
                                              const h16* __restrict__ W2t, const float* __restrict__ b2,
                                              const float* __restrict__ g2, const float* __restrict__ be2,
                                              float* __restrict__ out) {
  int b = blockIdx.x / m, c = blockIdx.x % m;
  const float* F = featIn + (size_t)b * n * 128;
  const float* P = posIn + (size_t)b * n * 3;
  const float* q = posQ + ((size_t)b * m + c) * 3;
  const int* nb = nbr + ((size_t)b * m + c) * 64;
  __shared__ h16 A0[64 * 160];
  __shared__ h16 A1[64 * 160];
  __shared__ int nbl[64];
  int t = threadIdx.x;
  if (t < 64) nbl[t] = nb[t];
  __syncthreads();
  float qx = q[0], qy = q[1], qz = q[2];
  for (int e = t; e < 64 * 160; e += 256) {
    int j = e / 160, k = e % 160;
    int g = nbl[j];
    int gi = g < 0 ? 0 : g;
    float v = 0.0f;
    if (k < 128)       v = F[(size_t)gi * 128 + k];
    else if (k == 128) v = P[3 * gi] - qx;
    else if (k == 129) v = P[3 * gi + 1] - qy;
    else if (k == 130) v = P[3 * gi + 2] - qz;
    A0[e] = (h16)v;
    A1[e] = (h16)0.0f;
  }
  __syncthreads();
  int wave = t >> 5, lane = t & 31;
  const float inv = rsqrtf(1.0f + 1e-5f);
  // GEMM1: [64x160] x [160x144]
  for (int nt = wave; nt < 9; nt += 8) {
    int ncol = nt * 16 + (lane & 15);
    float bb = ncol < 131 ? b1[ncol] : 0.0f;
    float gg = ncol < 131 ? g1[ncol] * inv : 0.0f;
    float ee = ncol < 131 ? be1[ncol] : 0.0f;
    for (int mt = 0; mt < 4; ++mt) {
      v8f acc = {};
#pragma unroll
      for (int kb = 0; kb < 5; ++kb) {
        v16h a = frag_a_lds(A0, 160, mt * 16, kb * 32, lane);
        v16h w = frag_b_glb(W1t, 160, nt * 16, kb * 32, lane);
        acc = wmma32(a, w, acc);
      }
      int mbase = mt * 16 + ((lane >> 4) << 3);
#pragma unroll
      for (int r = 0; r < 8; ++r) {
        float v = fmaxf(acc[r] + bb, 0.0f) * gg + ee;
        A1[(mbase + r) * 160 + ncol] = (h16)v;
      }
    }
  }
  __syncthreads();
  // GEMM2: [64x160] x [160x128] + masked column max over 64 rows
  {
    int nt = wave;                       // 8 waves, 8 N tiles
    int ncol = nt * 16 + (lane & 15);
    float bb = b2[ncol], gg = g2[ncol] * inv, ee = be2[ncol];
    float cmax = -1e30f;
    for (int mt = 0; mt < 4; ++mt) {
      v8f acc = {};
#pragma unroll
      for (int kb = 0; kb < 5; ++kb) {
        v16h a = frag_a_lds(A1, 160, mt * 16, kb * 32, lane);
        v16h w = frag_b_glb(W2t, 160, nt * 16, kb * 32, lane);
        acc = wmma32(a, w, acc);
      }
      int mbase = mt * 16 + ((lane >> 4) << 3);
#pragma unroll
      for (int r = 0; r < 8; ++r) {
        float v = fmaxf(acc[r] + bb, 0.0f) * gg + ee;
        if (nbl[mbase + r] < 0) v = -1e30f;
        cmax = fmaxf(cmax, v);
      }
    }
    cmax = fmaxf(cmax, __shfl_xor(cmax, 16, 32));
    if (lane < 16) out[((size_t)b * m + c) * 128 + ncol] = cmax;
  }
}

// ---------------- knn-3 inverse-d2 interpolation + concat skip -> [nF][256] -------------
__global__ void knn3_interp(const float* __restrict__ posF, int nF,
                            const float* __restrict__ posC, int nC,
                            const float* __restrict__ featC, const float* __restrict__ skip,
                            float* __restrict__ outIn, int total) {
  int gid = blockIdx.x * blockDim.x + threadIdx.x;
  if (gid >= total) return;
  int b = gid / nF, i = gid % nF;
  const float* pf = posF + ((size_t)b * nF + i) * 3;
  const float* PC = posC + (size_t)b * nC * 3;
  float px = pf[0], py = pf[1], pz = pf[2];
  float d0 = 1e30f, d1 = 1e30f, d2v = 1e30f;
  int i0 = 0, i1 = 0, i2 = 0;
  for (int j = 0; j < nC; ++j) {
    float dx = PC[3 * j] - px, dy = PC[3 * j + 1] - py, dz = PC[3 * j + 2] - pz;
    float d = dx * dx + dy * dy + dz * dz;
    if (d < d0)      { d2v = d1; i2 = i1; d1 = d0; i1 = i0; d0 = d; i0 = j; }
    else if (d < d1) { d2v = d1; i2 = i1; d1 = d; i1 = j; }
    else if (d < d2v){ d2v = d; i2 = j; }
  }
  float w0 = 1.0f / fmaxf(d0, 1e-16f);
  float w1 = 1.0f / fmaxf(d1, 1e-16f);
  float w2 = 1.0f / fmaxf(d2v, 1e-16f);
  float ws = w0 + w1 + w2;
  const float* f0 = featC + ((size_t)b * nC + i0) * 128;
  const float* f1 = featC + ((size_t)b * nC + i1) * 128;
  const float* f2 = featC + ((size_t)b * nC + i2) * 128;
  const float* sk = skip + ((size_t)b * nF + i) * 128;
  float* o = outIn + ((size_t)b * nF + i) * 256;
  for (int ch = 0; ch < 128; ++ch) {
    o[ch]       = (w0 * f0[ch] + w1 * f1[ch] + w2 * f2[ch]) / ws;
    o[128 + ch] = sk[ch];
  }
}

// ---------------- generic fused 2-layer MLP over dense rows (WMMA) ----------------------
// in[rows][Kin] f32 -> relu/bn -> relu/bn -> out[rows][N2r] f32. 32 rows per block.
__global__ __launch_bounds__(256) void mlp2_rows(const float* __restrict__ in, int rows,
                                                 int Kin, int KP,
                                                 const h16* __restrict__ W1t, int N1r, int N1P, int KP2,
                                                 const float* __restrict__ b1, const float* __restrict__ g1,
                                                 const float* __restrict__ be1,
                                                 const h16* __restrict__ W2t, int N2r, int N2P,
                                                 const float* __restrict__ b2, const float* __restrict__ g2,
                                                 const float* __restrict__ be2,
                                                 float* __restrict__ out) {
  extern __shared__ char smem[];
  h16* A0 = (h16*)smem;
  h16* A1 = A0 + 32 * KP;
  int t = threadIdx.x;
  int row0 = blockIdx.x * 32;
  for (int e = t; e < 32 * KP; e += 256) {
    int r = e / KP, k = e % KP;
    int rr = row0 + r;
    float v = (rr < rows && k < Kin) ? in[(size_t)rr * Kin + k] : 0.0f;
    A0[e] = (h16)v;
  }
  for (int e = t; e < 32 * KP2; e += 256) A1[e] = (h16)0.0f;
  __syncthreads();
  int wave = t >> 5, lane = t & 31;
  const float inv = rsqrtf(1.0f + 1e-5f);
  int NT1 = N1P / 16, NT2 = N2P / 16, KB1 = KP / 32, KB2 = KP2 / 32;
  for (int nt = wave; nt < NT1; nt += 8) {
    int ncol = nt * 16 + (lane & 15);
    float bb = ncol < N1r ? b1[ncol] : 0.0f;
    float gg = ncol < N1r ? g1[ncol] * inv : 0.0f;
    float ee = ncol < N1r ? be1[ncol] : 0.0f;
    for (int mt = 0; mt < 2; ++mt) {
      v8f acc = {};
      for (int kb = 0; kb < KB1; ++kb) {
        v16h a = frag_a_lds(A0, KP, mt * 16, kb * 32, lane);
        v16h w = frag_b_glb(W1t, KP, nt * 16, kb * 32, lane);
        acc = wmma32(a, w, acc);
      }
      int mbase = mt * 16 + ((lane >> 4) << 3);
#pragma unroll
      for (int r = 0; r < 8; ++r) {
        float v = fmaxf(acc[r] + bb, 0.0f) * gg + ee;
        A1[(mbase + r) * KP2 + ncol] = (h16)v;
      }
    }
  }
  __syncthreads();
  for (int nt = wave; nt < NT2; nt += 8) {
    int ncol = nt * 16 + (lane & 15);
    bool has = ncol < N2r;
    float bb = has ? b2[ncol] : 0.0f;
    float gg = has ? g2[ncol] * inv : 0.0f;
    float ee = has ? be2[ncol] : 0.0f;
    for (int mt = 0; mt < 2; ++mt) {
      v8f acc = {};
      for (int kb = 0; kb < KB2; ++kb) {
        v16h a = frag_a_lds(A1, KP2, mt * 16, kb * 32, lane);
        v16h w = frag_b_glb(W2t, KP2, nt * 16, kb * 32, lane);
        acc = wmma32(a, w, acc);
      }
      int mbase = mt * 16 + ((lane >> 4) << 3);
#pragma unroll
      for (int r = 0; r < 8; ++r) {
        int rr = row0 + mbase + r;
        float v = fmaxf(acc[r] + bb, 0.0f) * gg + ee;
        if (rr < rows && has) out[(size_t)rr * N2r + ncol] = v;
      }
    }
  }
}

// ======================================================================================
extern "C" void kernel_launch(void* const* d_in, const int* in_sizes, int n_in,
                              void* d_out, int out_size, void* d_ws, size_t ws_size,
                              hipStream_t stream) {
  (void)in_sizes; (void)n_in; (void)out_size; (void)ws_size;
  const float* x      = (const float*)d_in[0];
  const float* pos    = (const float*)d_in[1];
  const float* lin_w  = (const float*)d_in[3];
  const float* lin_b  = (const float*)d_in[4];
  const float* sa_w1  = (const float*)d_in[5];
  const float* sa_b1  = (const float*)d_in[6];
  const float* sa_g1  = (const float*)d_in[7];
  const float* sa_be1 = (const float*)d_in[8];
  const float* sa_w2  = (const float*)d_in[9];
  const float* sa_b2  = (const float*)d_in[10];
  const float* sa_g2  = (const float*)d_in[11];
  const float* sa_be2 = (const float*)d_in[12];
  const float* fp_w1  = (const float*)d_in[13];
  const float* fp_b1  = (const float*)d_in[14];
  const float* fp_g1  = (const float*)d_in[15];
  const float* fp_be1 = (const float*)d_in[16];
  const float* fp_w2  = (const float*)d_in[17];
  const float* fp_b2  = (const float*)d_in[18];
  const float* fp_g2  = (const float*)d_in[19];
  const float* fp_be2 = (const float*)d_in[20];
  const float* lo_w1  = (const float*)d_in[21];
  const float* lo_b1  = (const float*)d_in[22];
  const float* lo_g1  = (const float*)d_in[23];
  const float* lo_be1 = (const float*)d_in[24];
  const float* lo_w2  = (const float*)d_in[25];
  const float* lo_b2  = (const float*)d_in[26];
  const float* lo_g2  = (const float*)d_in[27];
  const float* lo_be2 = (const float*)d_in[28];

  char* ws = (char*)d_ws;
  size_t off = 0;
  auto alloc = [&](size_t bytes) -> char* {
    off = (off + 255) & ~(size_t)255;
    char* p = ws + off;
    off += bytes;
    return p;
  };

  float* xb   = (float*)alloc((size_t)4 * 4096 * 128 * 4);
  float* pos1 = (float*)alloc((size_t)4 * 2048 * 3 * 4);
  float* pos2 = (float*)alloc((size_t)4 * 1024 * 3 * 4);
  float* pos3 = (float*)alloc((size_t)4 * 512 * 3 * 4);
  int*   idx1 = (int*)alloc((size_t)4 * 2048 * 4);
  int*   idx2 = (int*)alloc((size_t)4 * 1024 * 4);
  int*   idx3 = (int*)alloc((size_t)4 * 512 * 4);
  int*   nbr1 = (int*)alloc((size_t)4 * 2048 * 64 * 4);
  int*   nbr2 = (int*)alloc((size_t)4 * 1024 * 64 * 4);
  int*   nbr3 = (int*)alloc((size_t)4 * 512 * 64 * 4);
  float* f1   = (float*)alloc((size_t)4 * 2048 * 128 * 4);
  float* f2   = (float*)alloc((size_t)4 * 1024 * 128 * 4);
  float* f3   = (float*)alloc((size_t)4 * 512 * 128 * 4);
  h16* saW1t[3], *saW2t[3], *fpW1t[3], *fpW2t[3];
  for (int i = 0; i < 3; ++i) {
    saW1t[i] = (h16*)alloc((size_t)144 * 160 * 2);
    saW2t[i] = (h16*)alloc((size_t)128 * 160 * 2);
    fpW1t[i] = (h16*)alloc((size_t)256 * 256 * 2);
    fpW2t[i] = (h16*)alloc((size_t)128 * 256 * 2);
  }
  h16* loW1t = (h16*)alloc((size_t)128 * 128 * 2);
  h16* loW2t = (h16*)alloc((size_t)16 * 128 * 2);
  float* fpin = (float*)alloc((size_t)4 * 4096 * 256 * 4);
  float* xfA  = (float*)alloc((size_t)4 * 4096 * 128 * 4);
  float* xfB  = (float*)alloc((size_t)4 * 4096 * 128 * 4);

  // ---- weight prep ----
  for (int i = 0; i < 3; ++i) {
    prep_wt<<<(144 * 160 + 255) / 256, 256, 0, stream>>>(sa_w1 + (size_t)i * 131 * 131, 131, 131, 160, 144, saW1t[i]);
    prep_wt<<<(128 * 160 + 255) / 256, 256, 0, stream>>>(sa_w2 + (size_t)i * 131 * 128, 131, 128, 160, 128, saW2t[i]);
    prep_wt<<<(256 * 256 + 255) / 256, 256, 0, stream>>>(fp_w1 + (size_t)i * 256 * 256, 256, 256, 256, 256, fpW1t[i]);
    prep_wt<<<(128 * 256 + 255) / 256, 256, 0, stream>>>(fp_w2 + (size_t)i * 256 * 128, 256, 128, 256, 128, fpW2t[i]);
  }
  prep_wt<<<(128 * 128 + 255) / 256, 256, 0, stream>>>(lo_w1, 128, 128, 128, 128, loW1t);
  prep_wt<<<(16 * 128 + 255) / 256, 256, 0, stream>>>(lo_w2, 128, 2, 128, 16, loW2t);

  // ---- lin_in ----
  lin_in_kernel<<<(16384 * 128 + 255) / 256, 256, 0, stream>>>(x, lin_w, lin_b, xb, 16384);

  // ---- SA pyramid ----
  const int npts[4] = {4096, 2048, 1024, 512};
  const float* posL[4] = {pos, pos1, pos2, pos3};
  float* posO[3] = {pos1, pos2, pos3};
  int* idxL[3] = {idx1, idx2, idx3};
  int* nbrL[3] = {nbr1, nbr2, nbr3};
  const float* featL[4] = {xb, f1, f2, f3};
  float* featO[3] = {f1, f2, f3};
  for (int i = 0; i < 3; ++i) {
    int n = npts[i], m = npts[i + 1];
    fps_kernel<<<4, 256, 0, stream>>>(posL[i], n, m, idxL[i], posO[i]);
    knn_ball<<<4 * m, 128, 0, stream>>>(posO[i], posL[i], m, n, nbrL[i]);
    sa_mlp<<<4 * m, 256, 0, stream>>>(featL[i], posL[i], posO[i], nbrL[i], m, n,
                                      saW1t[i], sa_b1 + (size_t)i * 131, sa_g1 + (size_t)i * 131, sa_be1 + (size_t)i * 131,
                                      saW2t[i], sa_b2 + (size_t)i * 128, sa_g2 + (size_t)i * 128, sa_be2 + (size_t)i * 128,
                                      featO[i]);
  }

  // ---- FP chain ----
  size_t shmFP = (size_t)32 * (256 + 256) * 2;
  // mi = 2: fine = level2 (1024), coarse feats = f3
  knn3_interp<<<(4 * 1024 + 255) / 256, 256, 0, stream>>>(pos2, 1024, pos3, 512, f3, f2, fpin, 4 * 1024);
  mlp2_rows<<<(4 * 1024 + 31) / 32, 256, shmFP, stream>>>(fpin, 4 * 1024, 256, 256,
      fpW1t[2], 256, 256, 256, fp_b1 + 2 * 256, fp_g1 + 2 * 256, fp_be1 + 2 * 256,
      fpW2t[2], 128, 128, fp_b2 + 2 * 128, fp_g2 + 2 * 128, fp_be2 + 2 * 128, xfA);
  // mi = 1: fine = level1 (2048)
  knn3_interp<<<(4 * 2048 + 255) / 256, 256, 0, stream>>>(pos1, 2048, pos2, 1024, xfA, f1, fpin, 4 * 2048);
  mlp2_rows<<<(4 * 2048 + 31) / 32, 256, shmFP, stream>>>(fpin, 4 * 2048, 256, 256,
      fpW1t[1], 256, 256, 256, fp_b1 + 1 * 256, fp_g1 + 1 * 256, fp_be1 + 1 * 256,
      fpW2t[1], 128, 128, fp_b2 + 1 * 128, fp_g2 + 1 * 128, fp_be2 + 1 * 128, xfB);
  // mi = 0: fine = level0 (4096), skip = xb
  knn3_interp<<<(4 * 4096 + 255) / 256, 256, 0, stream>>>(pos, 4096, pos1, 2048, xfB, xb, fpin, 4 * 4096);
  mlp2_rows<<<(4 * 4096 + 31) / 32, 256, shmFP, stream>>>(fpin, 4 * 4096, 256, 256,
      fpW1t[0], 256, 256, 256, fp_b1, fp_g1, fp_be1,
      fpW2t[0], 128, 128, fp_b2, fp_g2, fp_be2, xfA);

  // ---- output MLP: 128 -> 128 -> 2 ----
  size_t shmLO = (size_t)32 * (128 + 128) * 2;
  mlp2_rows<<<(4 * 4096 + 31) / 32, 256, shmLO, stream>>>(xfA, 4 * 4096, 128, 128,
      loW1t, 128, 128, 128, lo_b1, lo_g1, lo_be1,
      loW2t, 2, 16, lo_b2, lo_g2, lo_be2, (float*)d_out);
}
